// Fingerprint_MutiTask_87625922773464
// MI455X (gfx1250) — matrix operand order repaired and checked by reading, not compiled, not run on hardware
//
#include <hip/hip_runtime.h>
#include <hip/hip_bf16.h>
#include <math.h>

#define B_ 128
#define L_ 128
#define D_ 6
#define FD_ 256
#define NB_ 256
#define IN_ATOM_ 39
#define IN_BOND_ 10
#define M1_ (B_ * L_)        /* 16384 atom rows  */
#define M2_ (B_ * L_ * D_)   /* 98304 neighbor rows */
#define NEGV (-9e8f)

typedef __attribute__((ext_vector_type(16))) _Float16 v16h;
typedef __attribute__((ext_vector_type(8)))  _Float16 v8h;
typedef __attribute__((ext_vector_type(8)))  float    v8f;

__device__ __forceinline__ float lrelu_f(float x) { return x > 0.f ? x : 0.01f * x; }
__device__ __forceinline__ float elu_f(float x)   { return x > 0.f ? x : (expf(x) - 1.f); }
__device__ __forceinline__ float sigmoid_f(float x) { return 1.f / (1.f + expf(-x)); }

__device__ __forceinline__ float wave_sum(float v) {
#pragma unroll
    for (int o = 16; o > 0; o >>= 1) v += __shfl_xor(v, o, 32);
    return v;
}

// ---------------------------------------------------------------------------
// f32 -> f16 convert with zero-padding of the K dimension (also used to pack
// the atom_list GEMM A operand: rows x sk -> rows x dk, dk >= sk).
// ---------------------------------------------------------------------------
__global__ void cvt_pad_f16_kernel(const float* __restrict__ src, _Float16* __restrict__ dst,
                                   int rows, int sk, int dk) {
    int i = blockIdx.x * blockDim.x + threadIdx.x;
    int total = rows * dk;
    if (i >= total) return;
    int r = i / dk, c = i - r * dk;
    dst[i] = (c < sk) ? (_Float16)src[(size_t)r * sk + c] : (_Float16)0.f;
}

// ---------------------------------------------------------------------------
// Build the [M2, 64] f16 padded concat(atom_neighbor, bond_neighbor) rows.
// ---------------------------------------------------------------------------
__global__ void pack_nbr_kernel(const float* __restrict__ atom_list,
                                const float* __restrict__ bond_list,
                                const int* __restrict__ adl,
                                const int* __restrict__ bdl,
                                _Float16* __restrict__ dst) {
    size_t i = (size_t)blockIdx.x * blockDim.x + threadIdx.x;
    if (i >= (size_t)M2_ * 64) return;
    int r = (int)(i >> 6);
    int c = (int)(i & 63);
    int b = r / (L_ * D_);
    int rem = r - b * (L_ * D_);     // l*D + d
    float v = 0.f;
    if (c < IN_ATOM_) {
        int aidx = adl[(size_t)b * L_ * D_ + rem];
        v = atom_list[((size_t)b * L_ + aidx) * IN_ATOM_ + c];
    } else if (c < IN_ATOM_ + IN_BOND_) {
        int bidx = bdl[(size_t)b * L_ * D_ + rem];
        v = bond_list[((size_t)b * NB_ + bidx) * IN_BOND_ + (c - IN_ATOM_)];
    }
    dst[i] = (_Float16)v;
}

// ---------------------------------------------------------------------------
// Generic WMMA GEMM:  C[M,N] = A[M,K](f16,row-major) x W[N,K]^T(f16,row-major)
// 8 waves/block, each wave computes a 16x64 tile (A-frag reused across 4 WMMAs).
// mode 0: lrelu(acc+bias)   mode 1: elu(acc+rowscale[m]*bias)   mode 2: acc+bias
// ---------------------------------------------------------------------------
__global__ void wmma_gemm_kernel(const _Float16* __restrict__ A, int lda,
                                 const _Float16* __restrict__ W, int ldw,
                                 float* __restrict__ outF32,
                                 _Float16* __restrict__ outF16,
                                 const float* __restrict__ bias,
                                 const float* __restrict__ rowscale,
                                 int M, int N, int K, int mode) {
    const int lane = threadIdx.x & 31;
    const int wv = threadIdx.x >> 5;
    const int m0 = blockIdx.y * 128 + wv * 16;
    const int n0 = blockIdx.x * 64;
    if (m0 >= M) return;

    v8f acc[4] = {};

    const int mA = m0 + (lane & 15);
    const int koff = (lane >> 4) * 16;       // A-frag k-half select
    const int kcol = (lane >> 4) * 16;       // B-frag k-half select
    const _Float16* arow = A + (size_t)mA * lda;
    const int nB = n0 + (lane & 15);

    for (int kb = 0; kb < K; kb += 32) {
        v8h alo = *(const v8h*)(arow + kb + koff);
        v8h ahi = *(const v8h*)(arow + kb + 8 + koff);
        v16h af = __builtin_shufflevector(alo, ahi, 0, 1, 2, 3, 4, 5, 6, 7,
                                          8, 9, 10, 11, 12, 13, 14, 15);
#pragma unroll
        for (int j = 0; j < 4; j++) {
            const _Float16* wrow = W + (size_t)(nB + j * 16) * ldw + kb + kcol;
            v16h bf = *(const v16h*)wrow;
            acc[j] = __builtin_amdgcn_wmma_f32_16x16x32_f16(
                false, af, false, bf, (short)0, acc[j], false, false);
        }
    }

    const int mrow = m0 + ((lane >> 4) << 3);
#pragma unroll
    for (int j = 0; j < 4; j++) {
        const int n = n0 + j * 16 + (lane & 15);
        float bn = bias ? bias[n] : 0.f;
#pragma unroll
        for (int i = 0; i < 8; i++) {
            int m = mrow + i;
            float v = acc[j][i];
            if (mode == 1) v += rowscale[m] * bn; else v += bn;
            if (mode == 0) v = lrelu_f(v);
            else if (mode == 1) v = elu_f(v);
            if (outF32) outF32[(size_t)m * N + n] = v;
            if (outF16) outF16[(size_t)m * N + n] = (_Float16)v;
        }
    }
}

// ---------------------------------------------------------------------------
// Attention over D=6 neighbors: one wave per (b,l) row.
// Computes score -> masked softmax -> weighted neighbor sum (f16) + sum of
// weights (f32).  nbrGather=0: nbr row = (row*D+d) in nbrBase (r==0 path);
// nbrGather=1: nbr row = b*L + adl[row,d] in nbrBase (activated, r>=1 path).
// ---------------------------------------------------------------------------
__global__ void attn_score_kernel(const float* __restrict__ selfF,
                                  const _Float16* __restrict__ nbrBase,
                                  int nbrGather,
                                  const int* __restrict__ adl,
                                  const float* __restrict__ alignW,   // 512
                                  const float* __restrict__ alignBp,  // scalar
                                  _Float16* __restrict__ weighted,
                                  float* __restrict__ swOut) {
    const int lane = threadIdx.x & 31;
    const int row = blockIdx.x * (blockDim.x >> 5) + (threadIdx.x >> 5);
    if (row >= M1_) return;
    const int bmol = row >> 7;     // row / L
    const int f0 = lane * 8;
    const float ab = alignBp[0];

    float sf[8], ds = 0.f;
#pragma unroll
    for (int j = 0; j < 8; j++) {
        sf[j] = selfF[(size_t)row * FD_ + f0 + j];
        ds += sf[j] * alignW[f0 + j];
    }
    const float dot_self = wave_sum(ds);

    const _Float16* nrows[D_];
    float sc[D_];
    bool padd[D_];
#pragma unroll
    for (int d = 0; d < D_; d++) {
        int idx = adl[(size_t)row * D_ + d];
        padd[d] = (idx == L_ - 1);
        const _Float16* nr = nbrGather
            ? nbrBase + ((size_t)(bmol * L_ + idx)) * FD_
            : nbrBase + ((size_t)row * D_ + d) * FD_;
        nrows[d] = nr;
        float dn = 0.f;
#pragma unroll
        for (int j = 0; j < 8; j++) dn += (float)nr[f0 + j] * alignW[FD_ + f0 + j];
        dn = wave_sum(dn);
        sc[d] = lrelu_f(dot_self + dn + ab) + (padd[d] ? NEGV : 0.f);
    }

    float mx = sc[0];
#pragma unroll
    for (int d = 1; d < D_; d++) mx = fmaxf(mx, sc[d]);
    float den = 0.f, wgt[D_];
#pragma unroll
    for (int d = 0; d < D_; d++) { wgt[d] = expf(sc[d] - mx); den += wgt[d]; }
    float sw = 0.f;
#pragma unroll
    for (int d = 0; d < D_; d++) { wgt[d] = (wgt[d] / den) * (padd[d] ? 0.f : 1.f); sw += wgt[d]; }

    float wa[8] = {0, 0, 0, 0, 0, 0, 0, 0};
#pragma unroll
    for (int d = 0; d < D_; d++) {
        const _Float16* nr = nrows[d];
        float w = wgt[d];
#pragma unroll
        for (int j = 0; j < 8; j++) wa[j] += w * (float)nr[f0 + j];
    }
#pragma unroll
    for (int j = 0; j < 8; j++) weighted[(size_t)row * FD_ + f0 + j] = (_Float16)wa[j];
    if (lane == 0) swOut[row] = sw;
}

// ---------------------------------------------------------------------------
// GRU gating (elementwise) + relu; updates h in place, emits f16 mirrors.
// ---------------------------------------------------------------------------
__global__ void gru_elem_kernel(const float* __restrict__ gi, const float* __restrict__ gh,
                                float* __restrict__ hF32, _Float16* __restrict__ hF16,
                                float* __restrict__ actF32, _Float16* __restrict__ actF16) {
    size_t idx = (size_t)blockIdx.x * blockDim.x + threadIdx.x;
    if (idx >= (size_t)M1_ * FD_) return;
    size_t row = idx / FD_;
    int c = (int)(idx - row * FD_);
    size_t base = row * 768;
    float r = sigmoid_f(gi[base + c] + gh[base + c]);
    float z = sigmoid_f(gi[base + 256 + c] + gh[base + 256 + c]);
    float n = tanhf(gi[base + 512 + c] + r * gh[base + 512 + c]);
    float h = (1.f - z) * n + z * hF32[idx];
    hF32[idx] = h;
    hF16[idx] = (_Float16)h;
    float a = h > 0.f ? h : 0.f;
    actF32[idx] = a;
    actF16[idx] = (_Float16)a;
}

// ---------------------------------------------------------------------------
// mol_feature = sum_l activated*mask ; act_mol = relu(mol_feature)
// ---------------------------------------------------------------------------
__global__ void mol_sum_kernel(const float* __restrict__ actF32, const float* __restrict__ amask,
                               float* __restrict__ molFeat, float* __restrict__ actMol) {
    int b = blockIdx.x, f = threadIdx.x;
    float s = 0.f;
    for (int l = 0; l < L_; l++)
        s += actF32[((size_t)(b * L_ + l)) * FD_ + f] * amask[b * L_ + l];
    molFeat[b * FD_ + f] = s;
    actMol[b * FD_ + f] = s > 0.f ? s : 0.f;
}

// ---------------------------------------------------------------------------
// Mol attention: one block per molecule. score over L=128, masked softmax,
// mol_context = elu(sum_l w_l * act_t[l,:]).
// ---------------------------------------------------------------------------
__global__ void mol_attn_kernel(const float* __restrict__ actMol, const float* __restrict__ actF32,
                                const float* __restrict__ actT, const float* __restrict__ amask,
                                const float* __restrict__ mAlignW, const float* __restrict__ mAlignBp,
                                float* __restrict__ molCtx) {
    __shared__ float s_mol[FD_];
    __shared__ float s_aw[2 * FD_];
    __shared__ float s_s[L_];
    int b = blockIdx.x, t = threadIdx.x;
    s_mol[t] = actMol[b * FD_ + t];
    s_aw[t] = mAlignW[t];
    s_aw[t + FD_] = mAlignW[t + FD_];
    __syncthreads();
    if (t < L_) {
        const float* arow = actF32 + ((size_t)(b * L_ + t)) * FD_;
        float dot = 0.f;
        for (int f = 0; f < FD_; f++) dot += s_mol[f] * s_aw[f] + arow[f] * s_aw[FD_ + f];
        float mk = (amask[b * L_ + t] == 0.f) ? NEGV : 0.f;
        s_s[t] = lrelu_f(dot + mAlignBp[0]) + mk;
    }
    __syncthreads();
    if (t == 0) {
        float mx = s_s[0];
        for (int l = 1; l < L_; l++) mx = fmaxf(mx, s_s[l]);
        float den = 0.f;
        for (int l = 0; l < L_; l++) { s_s[l] = expf(s_s[l] - mx); den += s_s[l]; }
        float inv = 1.f / den;
        for (int l = 0; l < L_; l++) s_s[l] = s_s[l] * inv * amask[b * L_ + l];
    }
    __syncthreads();
    float acc = 0.f;
    for (int l = 0; l < L_; l++) acc += s_s[l] * actT[((size_t)(b * L_ + l)) * FD_ + t];
    molCtx[b * FD_ + t] = elu_f(acc);
}

// ---------------------------------------------------------------------------
// Mol GRU step: one block per molecule, full fp32 (tiny & precision-critical).
// Optionally writes relu(h) to the task output slice.
// ---------------------------------------------------------------------------
__global__ void mol_gru_kernel(const float* __restrict__ molCtx, float* __restrict__ molFeat,
                               float* __restrict__ actMol,
                               const float* __restrict__ wih, const float* __restrict__ whh,
                               const float* __restrict__ bih, const float* __restrict__ bhh,
                               float* __restrict__ taskOut) {
    __shared__ float xs[FD_], hs[FD_], sgi[3 * FD_], sgh[3 * FD_];
    int b = blockIdx.x, t = threadIdx.x;
    xs[t] = molCtx[b * FD_ + t];
    hs[t] = molFeat[b * FD_ + t];
    __syncthreads();
    for (int g = 0; g < 3; g++) {
        int c = g * FD_ + t;
        const float* wi = wih + (size_t)c * FD_;
        const float* wh = whh + (size_t)c * FD_;
        float di = 0.f, dh = 0.f;
        for (int k = 0; k < FD_; k++) { di += xs[k] * wi[k]; dh += hs[k] * wh[k]; }
        sgi[c] = di + bih[c];
        sgh[c] = dh + bhh[c];
    }
    __syncthreads();
    float r = sigmoid_f(sgi[t] + sgh[t]);
    float z = sigmoid_f(sgi[FD_ + t] + sgh[FD_ + t]);
    float n = tanhf(sgi[2 * FD_ + t] + r * sgh[2 * FD_ + t]);
    float h = (1.f - z) * n + z * hs[t];
    molFeat[b * FD_ + t] = h;
    float a = h > 0.f ? h : 0.f;
    actMol[b * FD_ + t] = a;
    if (taskOut) taskOut[b * FD_ + t] = a;
}

// ---------------------------------------------------------------------------
extern "C" void kernel_launch(void* const* d_in, const int* in_sizes, int n_in,
                              void* d_out, int out_size, void* d_ws, size_t ws_size,
                              hipStream_t stream) {
    (void)in_sizes; (void)n_in; (void)out_size; (void)ws_size;
    const float* atom_list   = (const float*)d_in[0];
    const float* bond_list   = (const float*)d_in[1];
    const int*   adl         = (const int*)d_in[2];
    const int*   bdl         = (const int*)d_in[3];
    const float* atom_mask   = (const float*)d_in[4];
    const float* atom_fc_w   = (const float*)d_in[5];
    const float* atom_fc_b   = (const float*)d_in[6];
    const float* nb_fc_w     = (const float*)d_in[7];
    const float* nb_fc_b     = (const float*)d_in[8];
    const float* align_w     = (const float*)d_in[9];    // [3,512]
    const float* align_b     = (const float*)d_in[10];   // [3]
    const float* attend_w    = (const float*)d_in[11];   // [3,256,256]
    const float* attend_b    = (const float*)d_in[12];   // [3,256]
    const float* gru_wih     = (const float*)d_in[13];   // [3,768,256]
    const float* gru_whh     = (const float*)d_in[14];
    const float* gru_bih     = (const float*)d_in[15];   // [3,768]
    const float* gru_bhh     = (const float*)d_in[16];
    const float* mgru_wih    = (const float*)d_in[17];   // [768,256]
    const float* mgru_whh    = (const float*)d_in[18];
    const float* mgru_bih    = (const float*)d_in[19];
    const float* mgru_bhh    = (const float*)d_in[20];
    const float* mol_align_w = (const float*)d_in[21];   // [4,512]
    const float* mol_align_b = (const float*)d_in[22];   // [4]
    const float* mol_att_w   = (const float*)d_in[23];   // [256,256]
    const float* mol_att_b   = (const float*)d_in[24];
    float* out = (float*)d_out;

    // ---- workspace layout ----
    char* base = (char*)d_ws;
    size_t off = 0;
    auto alloc = [&](size_t bytes) -> void* {
        void* p = base + off;
        off += (bytes + 255) & ~(size_t)255;
        return p;
    };
    _Float16* wAtomFc  = (_Float16*)alloc((size_t)256 * 64 * 2);
    _Float16* wNbFc    = (_Float16*)alloc((size_t)256 * 64 * 2);
    _Float16* wAttend  = (_Float16*)alloc((size_t)3 * 256 * 256 * 2);
    _Float16* wGruWih  = (_Float16*)alloc((size_t)3 * 768 * 256 * 2);
    _Float16* wGruWhh  = (_Float16*)alloc((size_t)3 * 768 * 256 * 2);
    _Float16* wMolAtt  = (_Float16*)alloc((size_t)256 * 256 * 2);
    _Float16* atomA    = (_Float16*)alloc((size_t)M1_ * 64 * 2);
    _Float16* nbrA     = (_Float16*)alloc((size_t)M2_ * 64 * 2);
    float*    hF32     = (float*)alloc((size_t)M1_ * FD_ * 4);   // also atom_feature
    _Float16* hF16     = (_Float16*)alloc((size_t)M1_ * FD_ * 2);
    _Float16* nbrF16   = (_Float16*)alloc((size_t)M2_ * FD_ * 2); // neighbor_feature (r==0)
    _Float16* wgtF16   = (_Float16*)alloc((size_t)M1_ * FD_ * 2);
    float*    swBuf    = (float*)alloc((size_t)M1_ * 4);
    _Float16* ctxF16   = (_Float16*)alloc((size_t)M1_ * FD_ * 2);
    float*    giF32    = (float*)alloc((size_t)M1_ * 768 * 4);
    float*    ghF32    = (float*)alloc((size_t)M1_ * 768 * 4);
    float*    actF32   = (float*)alloc((size_t)M1_ * FD_ * 4);
    _Float16* actF16   = (_Float16*)alloc((size_t)M1_ * FD_ * 2);
    float*    actT     = (float*)alloc((size_t)M1_ * FD_ * 4);
    float*    molFeat  = (float*)alloc((size_t)B_ * FD_ * 4);
    float*    actMol   = (float*)alloc((size_t)B_ * FD_ * 4);
    float*    molCtx   = (float*)alloc((size_t)B_ * FD_ * 4);

    auto cvt = [&](const float* src, _Float16* dst, int rows, int sk, int dk) {
        int total = rows * dk;
        cvt_pad_f16_kernel<<<(total + 255) / 256, 256, 0, stream>>>(src, dst, rows, sk, dk);
    };
    auto gemm = [&](const _Float16* A, int lda, const _Float16* W, int ldw,
                    float* oF, _Float16* oH, const float* bias, const float* rs,
                    int M, int N, int K, int mode) {
        dim3 g(N / 64, M / 128);
        wmma_gemm_kernel<<<g, 256, 0, stream>>>(A, lda, W, ldw, oF, oH, bias, rs, M, N, K, mode);
    };

    // ---- weight conversion + input packing ----
    cvt(atom_fc_w, wAtomFc, 256, IN_ATOM_, 64);
    cvt(nb_fc_w, wNbFc, 256, IN_ATOM_ + IN_BOND_, 64);
    cvt(attend_w, wAttend, 3 * 256, 256, 256);
    cvt(gru_wih, wGruWih, 3 * 768, 256, 256);
    cvt(gru_whh, wGruWhh, 3 * 768, 256, 256);
    cvt(mol_att_w, wMolAtt, 256, 256, 256);
    cvt(atom_list, atomA, M1_, IN_ATOM_, 64);
    {
        size_t total = (size_t)M2_ * 64;
        pack_nbr_kernel<<<(unsigned)((total + 255) / 256), 256, 0, stream>>>(
            atom_list, bond_list, adl, bdl, nbrA);
    }

    // ---- atom_fc: atom_feature = lrelu(...)  (f32 -> hF32, f16 -> hF16 = initial h) ----
    gemm(atomA, 64, wAtomFc, 64, hF32, hF16, atom_fc_b, nullptr, M1_, FD_, 64, 0);
    // ---- neighbor_fc: neighbor_feature (f16 only) ----
    gemm(nbrA, 64, wNbFc, 64, nullptr, nbrF16, nb_fc_b, nullptr, M2_, FD_, 64, 0);

    // ---- radius layers ----
    for (int r = 0; r < 3; r++) {
        const float* selfF = (r == 0) ? hF32 : actF32;
        const _Float16* nbrB = (r == 0) ? nbrF16 : actF16;
        attn_score_kernel<<<M1_ / 4, 128, 0, stream>>>(
            selfF, nbrB, (r == 0) ? 0 : 1, adl,
            align_w + (size_t)r * 512, align_b + r, wgtF16, swBuf);
        // context = elu(weighted @ attend_w^T + sw*attend_b)  -> f16
        gemm(wgtF16, FD_, wAttend + (size_t)r * 256 * 256, FD_,
             nullptr, ctxF16, attend_b + (size_t)r * 256, swBuf, M1_, FD_, FD_, 1);
        // GRU gemms
        gemm(ctxF16, FD_, wGruWih + (size_t)r * 768 * 256, FD_,
             giF32, nullptr, gru_bih + (size_t)r * 768, nullptr, M1_, 768, FD_, 2);
        gemm(hF16, FD_, wGruWhh + (size_t)r * 768 * 256, FD_,
             ghF32, nullptr, gru_bhh + (size_t)r * 768, nullptr, M1_, 768, FD_, 2);
        gru_elem_kernel<<<(M1_ * FD_) / 256, 256, 0, stream>>>(
            giF32, ghF32, hF32, hF16, actF32, actF16);
    }

    // ---- molecular phase ----
    mol_sum_kernel<<<B_, FD_, 0, stream>>>(actF32, atom_mask, molFeat, actMol);
    gemm(actF16, FD_, wMolAtt, FD_, actT, nullptr, mol_att_b, nullptr, M1_, FD_, FD_, 2);

    for (int i = 0; i < 4; i++) {
        for (int t = 0; t < 2; t++) {
            mol_attn_kernel<<<B_, FD_, 0, stream>>>(
                actMol, actF32, actT, atom_mask,
                mol_align_w + (size_t)i * 512, mol_align_b + i, molCtx);
            float* taskOut = (t == 1) ? (out + (size_t)i * B_ * FD_) : nullptr;
            mol_gru_kernel<<<B_, FD_, 0, stream>>>(
                molCtx, molFeat, actMol, mgru_wih, mgru_whh, mgru_bih, mgru_bhh, taskOut);
        }
    }
}
